// DecoderModule_41223096107032
// MI455X (gfx1250) — matrix operand order repaired and checked by reading, not compiled, not run on hardware
//
#include <hip/hip_runtime.h>
#include <hip/hip_bf16.h>
#include <cstdint>

#define E_DIM 768
#define H_HEADS 12
#define N_TOK 4096
#define FF_DIM 3072
#define HD_DIM 64
#define LN_EPS 1e-5f

typedef __attribute__((ext_vector_type(16))) _Float16 v16h;
typedef __attribute__((ext_vector_type(8)))  float    v8f;

union AFrag { v16h v; uint32_t u[8]; };

__device__ __forceinline__ v8f wmma16(const AFrag& a, const AFrag& b, v8f c) {
  // D = A(16x32 f16) * B(32x16 f16) + C(16x16 f32)
  return __builtin_amdgcn_wmma_f32_16x16x32_f16(false, a.v, false, b.v,
                                                (short)0, c, false, false);
}

__device__ __forceinline__ v8f zero8() {
  v8f z = {0.f,0.f,0.f,0.f,0.f,0.f,0.f,0.f};
  return z;
}

// ---------------------------------------------------------------- casts
__global__ __launch_bounds__(256)
void cast_f16_kernel(const float* __restrict__ in, _Float16* __restrict__ out, int n) {
  int i = blockIdx.x * 256 + threadIdx.x;
  if (i < n) out[i] = (_Float16)in[i];
}

// in: K x N (row-major f32)  ->  out: N x K (row-major f16)
__global__ __launch_bounds__(256)
void transpose_f16_kernel(const float* __restrict__ in, _Float16* __restrict__ out,
                          int K, int N) {
  int i = blockIdx.x * 256 + threadIdx.x;
  if (i < K * N) {
    int n = i / K;
    int k = i - n * K;
    out[i] = (_Float16)in[(size_t)k * N + n];
  }
}

// ---------------------------------------------------------------- GEMM
// C(MxN) = A(MxK f16, row-major) * BT(NxK f16, row-major == B column-major)
// B tile is staged into LDS with gfx1250 async global->LDS DMA, double-buffered.
// STORE_MODE: 0 = f32 flat, 1 = f16 flat, 2 = f16 with per-head V transpose remap
template<int STORE_MODE, bool BIAS, bool RELU>
__global__ __launch_bounds__(256)
void gemm_f16_kernel(const _Float16* __restrict__ A,
                     const _Float16* __restrict__ BT,
                     const float*    __restrict__ bias,
                     float*          __restrict__ Cf32,
                     _Float16*       __restrict__ Cf16,
                     int M, int Ncols, int K) {
  __shared__ _Float16 Bs[2][64 * 32];       // double-buffered 64 cols x 32 k
  const int lane = threadIdx.x & 31;
  const int wave = threadIdx.x >> 5;        // 0..7
  const int grp  = lane >> 4;               // lane half
  const int ln16 = lane & 15;
  const int colBase = blockIdx.x * 64;
  const int rowBase = blockIdx.y * 128 + wave * 16;

  // async-copy lane assignment: 16B per thread, 4KB per stage
  const int ldr = threadIdx.x >> 2;         // 0..63 : output column (row of BT)
  const int ldc = (threadIdx.x & 3) * 8;    // 0,8,16,24 : halfs within k-window
  const _Float16* bsrc = BT + (size_t)(colBase + ldr) * K + ldc;

  v8f acc[4];
  #pragma unroll
  for (int j = 0; j < 4; ++j) acc[j] = zero8();

  // issue one async 16B global->LDS copy per thread for stage (buf, k0)
  auto issue_b = [&](int buf, int k0) {
    // low 32 bits of the flat shared-memory address == hardware LDS address
    uint32_t lds_addr = (uint32_t)(uintptr_t)(&Bs[buf][ldr * 32 + ldc]);
    uint64_t gaddr    = (uint64_t)(uintptr_t)(bsrc + k0);
    asm volatile("global_load_async_to_lds_b128 %0, %1, off"
                 :: "v"(lds_addr), "v"(gaddr) : "memory");
  };

  const int nsteps = K / 32;
  issue_b(0, 0);                            // prime the pipeline

  for (int i = 0; i < nsteps; ++i) {
    const int buf = i & 1;
    const int k0  = i * 32;
    if (i + 1 < nsteps) {
      issue_b(buf ^ 1, k0 + 32);            // prefetch next stage
      asm volatile("s_wait_asynccnt 0x1" ::: "memory");  // stage i landed
    } else {
      asm volatile("s_wait_asynccnt 0x0" ::: "memory");
    }
    __syncthreads();                        // all waves' DMA for stage i done

    // A fragment: lane = row M, contiguous k = 16*half + s, same order as B
    AFrag a;
    {
      const _Float16* arow = A + (size_t)(rowBase + ln16) * K + k0;
      __builtin_prefetch((const void*)(arow + 32), 0, 3);
      #pragma unroll
      for (int p = 0; p < 8; ++p) {
        int kk = grp * 16 + p * 2;
        a.u[p] = *(const uint32_t*)(arow + kk);
      }
    }
    #pragma unroll
    for (int j = 0; j < 4; ++j) {
      AFrag b;  // B 32x16, contiguous k = 16*half + s: lane=N col, K pairs per VGPR
      const _Float16* brow = &Bs[buf][(j * 16 + ln16) * 32 + grp * 16];
      #pragma unroll
      for (int p = 0; p < 8; ++p) b.u[p] = *(const uint32_t*)(brow + p * 2);
      acc[j] = wmma16(a, b, acc[j]);
    }
    __syncthreads();                        // done reading buf before overwrite
  }

  // epilogue: C-layout lane L, vgpr r -> (M = r + 8*(L/16), N = L%16)
  #pragma unroll
  for (int j = 0; j < 4; ++j) {
    int col = colBase + j * 16 + ln16;
    float bv = BIAS ? bias[col] : 0.f;
    #pragma unroll
    for (int r = 0; r < 8; ++r) {
      int row = rowBase + r + grp * 8;
      float v = acc[j][r] + bv;
      if (RELU) v = fmaxf(v, 0.f);
      size_t flat = (size_t)row * Ncols + col;
      if (STORE_MODE == 0) {
        Cf32[flat] = v;
      } else if (STORE_MODE == 1) {
        Cf16[flat] = (_Float16)v;
      } else {
        // reference reshape(H, N, HD) on the *flat* array, then transpose per
        // head so P*V B-frags read contiguous keys: Vt[h][d][n]
        size_t h  = flat >> 18;            // / (N_TOK*HD_DIM)
        size_t rm = flat & 262143u;
        size_t nk = rm >> 6;
        size_t d  = rm & 63u;
        Cf16[(h << 18) + d * (size_t)N_TOK + nk] = (_Float16)v;
      }
    }
  }
}

// ---------------------------------------------------------------- attention
// One wave per (16-query block, head). Flash-style online softmax.
__global__ __launch_bounds__(32)
void attn_kernel(const _Float16* __restrict__ Qb,
                 const _Float16* __restrict__ Kb,
                 const _Float16* __restrict__ Vt,
                 _Float16*       __restrict__ Ctx) {
  __shared__ _Float16 Ps[16 * 32];
  const int lane = threadIdx.x & 31;
  const int grp  = lane >> 4;
  const int ln16 = lane & 15;
  const int qb = blockIdx.x;                 // 0..255
  const int h  = blockIdx.y;                 // 0..11
  const size_t hb = (size_t)h << 18;         // h * N_TOK * HD_DIM
  const int q0 = qb * 16;

  // Q fragments over head-dim halves (A 16x32 layout)
  AFrag aq[2];
  {
    const _Float16* qrow = Qb + hb + (size_t)(q0 + ln16) * HD_DIM;
    #pragma unroll
    for (int hf = 0; hf < 2; ++hf)
      #pragma unroll
      for (int p = 0; p < 8; ++p) {
        int kk = hf * 32 + grp * 16 + p * 2;
        aq[hf].u[p] = *(const uint32_t*)(qrow + kk);
      }
  }

  v8f O[4];
  #pragma unroll
  for (int j = 0; j < 4; ++j) O[j] = zero8();
  float mrun[8], lrun[8];
  #pragma unroll
  for (int r = 0; r < 8; ++r) { mrun[r] = -1e30f; lrun[r] = 0.f; }

  const int kbmax = qb >> 1;
  for (int kb2 = 0; kb2 <= kbmax; ++kb2) {
    const int kbase = kb2 * 32;
    // ---- scores S(16q x 32k), two 16x16 tiles, contraction over hd=64
    v8f s[2];
    #pragma unroll
    for (int t = 0; t < 2; ++t) {
      v8f a0 = zero8();
      #pragma unroll
      for (int hf = 0; hf < 2; ++hf) {
        AFrag bk;  // B 32x16: lane = key, K = hd pairs (contiguous in Kb)
        const _Float16* krow =
            Kb + hb + (size_t)(kbase + t * 16 + ln16) * HD_DIM + hf * 32 + grp * 16;
        #pragma unroll
        for (int p = 0; p < 8; ++p) bk.u[p] = *(const uint32_t*)(krow + p * 2);
        a0 = wmma16(aq[hf], bk, a0);
      }
      s[t] = a0;
    }
    // ---- scale + causal mask (diagonal step only)
    const bool diag = (kb2 == kbmax);
    #pragma unroll
    for (int t = 0; t < 2; ++t)
      #pragma unroll
      for (int r = 0; r < 8; ++r) {
        float v = s[t][r] * 0.125f;          // 1/sqrt(64)
        if (diag) {
          int key = kbase + t * 16 + ln16;
          int qy  = q0 + r + grp * 8;
          if (key > qy) v = -1e30f;
        }
        s[t][r] = v;
      }
    // ---- online softmax (rows span 16 lanes of a half-wave)
    float mnew[8], alpha[8];
    #pragma unroll
    for (int r = 0; r < 8; ++r) {
      float mx = fmaxf(s[0][r], s[1][r]);
      #pragma unroll
      for (int d = 1; d <= 8; d <<= 1) mx = fmaxf(mx, __shfl_xor(mx, d, 32));
      mnew[r]  = fmaxf(mrun[r], mx);
      alpha[r] = __expf(mrun[r] - mnew[r]);
      mrun[r]  = mnew[r];
    }
    #pragma unroll
    for (int t = 0; t < 2; ++t)
      #pragma unroll
      for (int r = 0; r < 8; ++r) s[t][r] = __expf(s[t][r] - mnew[r]);
    #pragma unroll
    for (int r = 0; r < 8; ++r) {
      float sm = s[0][r] + s[1][r];
      #pragma unroll
      for (int d = 1; d <= 8; d <<= 1) sm += __shfl_xor(sm, d, 32);
      lrun[r] = lrun[r] * alpha[r] + sm;
    }
    #pragma unroll
    for (int j = 0; j < 4; ++j)
      #pragma unroll
      for (int r = 0; r < 8; ++r) O[j][r] *= alpha[r];

    // ---- P (C layout) -> LDS row-major -> A fragment (16x32)
    #pragma unroll
    for (int t = 0; t < 2; ++t)
      #pragma unroll
      for (int r = 0; r < 8; ++r)
        Ps[(r + 8 * grp) * 32 + t * 16 + ln16] = (_Float16)s[t][r];
    __syncthreads();
    AFrag pa;
    {
      const _Float16* prow = &Ps[ln16 * 32];
      #pragma unroll
      for (int p = 0; p < 8; ++p) {
        int kk = grp * 16 + p * 2;
        pa.u[p] = *(const uint32_t*)(prow + kk);
      }
    }
    // ---- ctx += P(16x32) * V(32k x 64hd); Vt[h][d][keys] contiguous keys
    #pragma unroll
    for (int j = 0; j < 4; ++j) {
      AFrag bv;
      const _Float16* vrow =
          Vt + hb + (size_t)(j * 16 + ln16) * N_TOK + kbase + grp * 16;
      #pragma unroll
      for (int p = 0; p < 8; ++p) bv.u[p] = *(const uint32_t*)(vrow + p * 2);
      O[j] = wmma16(pa, bv, O[j]);
    }
    __syncthreads();   // protect Ps before next iteration overwrites it
  }

  // epilogue: write ctx in reference merge layout tok*E + h*64 + d
  #pragma unroll
  for (int j = 0; j < 4; ++j)
    #pragma unroll
    for (int r = 0; r < 8; ++r) {
      int row = q0 + r + grp * 8;
      int col = h * HD_DIM + j * 16 + ln16;
      Ctx[(size_t)row * E_DIM + col] = (_Float16)(O[j][r] / lrun[r]);
    }
}

// ---------------------------------------------------------------- LayerNorm
// out = LN(X + R) * g + b ; writes f32 (always) and f16 (if outH != null)
__global__ __launch_bounds__(256)
void ln_residual_kernel(const float* __restrict__ X, const float* __restrict__ R,
                        const float* __restrict__ g, const float* __restrict__ b,
                        float* __restrict__ outF, _Float16* __restrict__ outH) {
  __shared__ float sred[8];
  const int row = blockIdx.x;
  const int tid = threadIdx.x;
  float v0[3];
  float s = 0.f;
  #pragma unroll
  for (int i = 0; i < 3; ++i) {
    int c = tid + i * 256;
    float t = X[(size_t)row * E_DIM + c] + R[(size_t)row * E_DIM + c];
    v0[i] = t; s += t;
  }
  #pragma unroll
  for (int d = 16; d >= 1; d >>= 1) s += __shfl_xor(s, d, 32);
  if ((tid & 31) == 0) sred[tid >> 5] = s;
  __syncthreads();
  float tot = 0.f;
  #pragma unroll
  for (int i = 0; i < 8; ++i) tot += sred[i];
  __syncthreads();
  float mean = tot * (1.f / E_DIM);

  float vs = 0.f;
  #pragma unroll
  for (int i = 0; i < 3; ++i) { float dd = v0[i] - mean; vs += dd * dd; }
  #pragma unroll
  for (int d = 16; d >= 1; d >>= 1) vs += __shfl_xor(vs, d, 32);
  if ((tid & 31) == 0) sred[tid >> 5] = vs;
  __syncthreads();
  float vtot = 0.f;
  #pragma unroll
  for (int i = 0; i < 8; ++i) vtot += sred[i];
  float rstd = rsqrtf(vtot * (1.f / E_DIM) + LN_EPS);

  #pragma unroll
  for (int i = 0; i < 3; ++i) {
    int c = tid + i * 256;
    float o = (v0[i] - mean) * rstd * g[c] + b[c];
    outF[(size_t)row * E_DIM + c] = o;
    if (outH) outH[(size_t)row * E_DIM + c] = (_Float16)o;
  }
}

// ---------------------------------------------------------------- launch
extern "C" void kernel_launch(void* const* d_in, const int* in_sizes, int n_in,
                              void* d_out, int out_size, void* d_ws, size_t ws_size,
                              hipStream_t stream) {
  const float* x     = (const float*)d_in[0];
  const float* Wq    = (const float*)d_in[1];
  const float* Wk    = (const float*)d_in[2];
  const float* Wv    = (const float*)d_in[3];
  const float* Wo    = (const float*)d_in[4];
  const float* W1    = (const float*)d_in[5];
  const float* b1    = (const float*)d_in[6];
  const float* W2    = (const float*)d_in[7];
  const float* b2    = (const float*)d_in[8];
  const float* g1    = (const float*)d_in[9];
  const float* beta1 = (const float*)d_in[10];
  const float* g2    = (const float*)d_in[11];
  const float* beta2 = (const float*)d_in[12];

  const size_t NE = (size_t)N_TOK * E_DIM;      // 3,145,728
  const size_t EE = (size_t)E_DIM * E_DIM;
  const size_t EF = (size_t)E_DIM * FF_DIM;
  const size_t NF = (size_t)N_TOK * FF_DIM;

  char* p = (char*)d_ws;
  auto alloc = [&](size_t bytes) -> void* {
    void* r = (void*)p;
    p += (bytes + 255) & ~(size_t)255;
    return r;
  };
  _Float16* Xb   = (_Float16*)alloc(NE * 2);
  _Float16* WqT  = (_Float16*)alloc(EE * 2);
  _Float16* WkT  = (_Float16*)alloc(EE * 2);
  _Float16* WvT  = (_Float16*)alloc(EE * 2);
  _Float16* WoT  = (_Float16*)alloc(EE * 2);
  _Float16* W1T  = (_Float16*)alloc(EF * 2);
  _Float16* W2T  = (_Float16*)alloc(EF * 2);
  _Float16* Qb   = (_Float16*)alloc(NE * 2);
  _Float16* Kb   = (_Float16*)alloc(NE * 2);
  _Float16* Vt   = (_Float16*)alloc(NE * 2);
  _Float16* Ctx  = (_Float16*)alloc(NE * 2);
  float*    attn = (float*)   alloc(NE * 4);
  float*    h1   = (float*)   alloc(NE * 4);
  _Float16* H1b  = (_Float16*)alloc(NE * 2);
  _Float16* F1b  = (_Float16*)alloc(NF * 2);
  float*    ff2  = (float*)   alloc(NE * 4);

  // casts / transposes
  cast_f16_kernel<<<(int)(NE / 256), 256, 0, stream>>>(x, Xb, (int)NE);
  transpose_f16_kernel<<<(int)((EE + 255) / 256), 256, 0, stream>>>(Wq, WqT, E_DIM, E_DIM);
  transpose_f16_kernel<<<(int)((EE + 255) / 256), 256, 0, stream>>>(Wk, WkT, E_DIM, E_DIM);
  transpose_f16_kernel<<<(int)((EE + 255) / 256), 256, 0, stream>>>(Wv, WvT, E_DIM, E_DIM);
  transpose_f16_kernel<<<(int)((EE + 255) / 256), 256, 0, stream>>>(Wo, WoT, E_DIM, E_DIM);
  transpose_f16_kernel<<<(int)((EF + 255) / 256), 256, 0, stream>>>(W1, W1T, E_DIM, FF_DIM);
  transpose_f16_kernel<<<(int)((EF + 255) / 256), 256, 0, stream>>>(W2, W2T, FF_DIM, E_DIM);

  dim3 gE(E_DIM / 64, N_TOK / 128);     // 12 x 32
  dim3 gF(FF_DIM / 64, N_TOK / 128);    // 48 x 32

  // QKV projections (f16 outputs; V transposed per head)
  gemm_f16_kernel<1, false, false><<<gE, 256, 0, stream>>>(Xb, WqT, nullptr, nullptr, Qb,  N_TOK, E_DIM, E_DIM);
  gemm_f16_kernel<1, false, false><<<gE, 256, 0, stream>>>(Xb, WkT, nullptr, nullptr, Kb,  N_TOK, E_DIM, E_DIM);
  gemm_f16_kernel<2, false, false><<<gE, 256, 0, stream>>>(Xb, WvT, nullptr, nullptr, Vt,  N_TOK, E_DIM, E_DIM);

  // causal flash attention
  attn_kernel<<<dim3(N_TOK / 16, H_HEADS), 32, 0, stream>>>(Qb, Kb, Vt, Ctx);

  // output projection + residual LN1
  gemm_f16_kernel<0, false, false><<<gE, 256, 0, stream>>>(Ctx, WoT, nullptr, attn, nullptr, N_TOK, E_DIM, E_DIM);
  ln_residual_kernel<<<N_TOK, 256, 0, stream>>>(x, attn, g1, beta1, h1, H1b);

  // FFN
  gemm_f16_kernel<1, true, true ><<<gF, 256, 0, stream>>>(H1b, W1T, b1, nullptr, F1b, N_TOK, FF_DIM, E_DIM);
  gemm_f16_kernel<0, true, false><<<gE, 256, 0, stream>>>(F1b, W2T, b2, ff2, nullptr, N_TOK, E_DIM, FF_DIM);

  // residual LN2 -> output
  ln_residual_kernel<<<N_TOK, 256, 0, stream>>>(h1, ff2, g2, beta2, (float*)d_out, nullptr);

  (void)in_sizes; (void)n_in; (void)out_size; (void)ws_size;
}
